// RAVENoiseGenerator_1477468750589
// MI455X (gfx1250) — matrix operand order repaired and checked by reading, not compiled, not run on hardware
//
#include <hip/hip_runtime.h>
#include <hip/hip_bf16.h>

typedef __attribute__((ext_vector_type(16))) __bf16 v16bf;
typedef __attribute__((ext_vector_type(8)))  __bf16 v8bf;
typedef __attribute__((ext_vector_type(8)))  float  v8f;

// ---------------------------------------------------------------------------
// Weight prep: w[Cout][Cin][3] fp32 -> wbf[3][Cout][Cin] bf16 (tap-major, so a
// K-chunk weight tile is rows of 32 contiguous bf16 = async-b128 copyable).
// ---------------------------------------------------------------------------
__global__ __launch_bounds__(256) void weight_prep_kernel(
    const float* __restrict__ w, __bf16* __restrict__ wbf, int Cout, int Cin) {
  int idx = blockIdx.x * 256 + threadIdx.x;   // (co, ci) pair
  if (idx >= Cout * Cin) return;
  int co = idx / Cin, ci = idx % Cin;
  const float* wp = w + (size_t)idx * 3;
#pragma unroll
  for (int tap = 0; tap < 3; ++tap)
    wbf[((size_t)tap * Cout + co) * (size_t)Cin + ci] = (__bf16)wp[tap];
}

// ---------------------------------------------------------------------------
// Implicit-GEMM strided conv1d (K=3, stride=2, pad=(0,1)) using bf16 WMMA.
//   in  : [B][Cin][Tin]      (float or bf16)
//   wbf : [3][Cout][Cin]     (bf16, pre-transposed)
//   out : [B][Cout][Tout]    (bf16), Tout = Tin/2
// Workgroup tile: 128 Cout x 64 T, 8 waves, each wave 32x32 via 2x2 WMMA tiles.
// Weights staged to LDS with GLOBAL_LOAD_ASYNC_TO_LDS_B128 (ASYNCcnt), which
// overlaps the VALU transpose/convert staging of the input slab.
// ---------------------------------------------------------------------------
template <typename TIN, bool RELU>
__global__ __launch_bounds__(256) void conv_wmma_kernel(
    const TIN* __restrict__ in, const __bf16* __restrict__ wbf,
    const float* __restrict__ bias, __bf16* __restrict__ out,
    int Cin, int Cout, int Tin, int Tout) {
  __shared__ __align__(16) __bf16 sW[3][128][32];   // [tap][cout][cin]  24 KB
  __shared__ __align__(16) __bf16 sX[132][32];      // [t_in - 2*t0][cin] 8.25 KB

  const int tid   = threadIdx.x;
  const int b     = blockIdx.z;
  const int coBlk = blockIdx.y * 128;
  const int t0    = blockIdx.x * 64;

  const int wave   = tid >> 5;
  const int lane   = tid & 31;
  const int wm     = wave & 3;    // 0..3 : cout sub-tile (4*32 = 128)
  const int wn     = wave >> 2;   // 0..1 : time sub-tile (2*32 = 64)
  const int laneHi = lane >> 4;   // half-wave select
  const int lane15 = lane & 15;

  // LDS byte offset of sW (generic LDS pointer: low 32 bits = LDS offset)
  const unsigned ldsW = (unsigned)(uintptr_t)(&sW[0][0][0]);

  v8f acc[2][2] = {};

  for (int c0 = 0; c0 < Cin; c0 += 32) {
    __syncthreads();
    // ---- stage weights via async DMA: wbf[tap][coBlk+col][c0 .. c0+32)
    // 1536 x 16B transfers, 6 per thread; LDS dest is linear (tap,col,q).
#pragma unroll
    for (int it = 0; it < 6; ++it) {
      int idx = tid + it * 256;          // 0..1535
      int row = idx >> 2;                // tap*128 + col
      int q   = idx & 3;                 // 8-element quarter of the 32-cin row
      int tap = row >> 7;
      int col = row & 127;
      unsigned goff = (unsigned)((((size_t)tap * Cout + coBlk + col) * Cin +
                                  c0 + q * 8) * sizeof(__bf16));
      unsigned loff = ldsW + (unsigned)idx * 16u;
      asm volatile("global_load_async_to_lds_b128 %0, %1, %2"
                   :: "v"(loff), "v"(goff), "s"(wbf)
                   : "memory");
    }
    // ---- stage input slab transposed (VALU: convert + transpose + zero-pad):
    // sX[d][ci] = in[b][c0+ci][2*t0+d]
    for (int idx = tid; idx < 32 * 132; idx += 256) {
      int ci = idx / 132;
      int d  = idx % 132;
      int tg = 2 * t0 + d;
      float v = 0.0f;
      if (d < 130 && tg < Tin)
        v = (float)in[((size_t)b * Cin + (c0 + ci)) * (size_t)Tin + tg];
      sX[d][ci] = (__bf16)v;
    }
    // prefetch next input chunk into L2 (global_prefetch_b8)
    if (c0 + 32 < Cin) {
      __builtin_prefetch(
          in + ((size_t)b * Cin + (c0 + 32 + (tid & 31))) * (size_t)Tin +
              2 * t0 + (tid >> 5) * 16, 0, 1);
    }
    // wave-local completion of this wave's async copies, then publish via barrier
    asm volatile("s_wait_asynccnt 0x0" ::: "memory");
    __syncthreads();

    const int coL = wm * 32;
    const int tL  = wn * 32;
#pragma unroll
    for (int tap = 0; tap < 3; ++tap) {
      // A fragments (16x32 bf16, M=cout, K=cin chunk), ISA wave32 layout:
      // lanes 0-15: M=lane, K = 0..7 & 16..23 ; lanes 16-31: M=lane-16, K = 8..15 & 24..31
      v16bf afrag[2];
#pragma unroll
      for (int mi = 0; mi < 2; ++mi) {
        const __bf16* ap = &sW[tap][coL + mi * 16 + lane15][laneHi ? 8 : 0];
        union { v16bf v; v8bf h[2]; } u;
        u.h[0] = *(const v8bf*)ap;         // 16B aligned ds_load_b128
        u.h[1] = *(const v8bf*)(ap + 16);
        afrag[mi] = u.v;
      }
      // B fragments (32x16 bf16, K=cin, N=t): lanes 0-15 carry K=0..7 & 16..23,
      // lanes 16-31 carry K=8..15 & 24..31, column N = lane%16.
      v16bf bfrag[2];
#pragma unroll
      for (int ni = 0; ni < 2; ++ni) {
        int tl = tL + ni * 16 + lane15;          // local output time
        const __bf16* bp = &sX[2 * tl + tap][laneHi ? 8 : 0];
        union { v16bf v; v8bf h[2]; } u;
        u.h[0] = *(const v8bf*)bp;
        u.h[1] = *(const v8bf*)(bp + 16);
        bfrag[ni] = u.v;
      }
#pragma unroll
      for (int mi = 0; mi < 2; ++mi)
#pragma unroll
        for (int ni = 0; ni < 2; ++ni)
          acc[mi][ni] = __builtin_amdgcn_wmma_f32_16x16x32_bf16(
              false, afrag[mi], false, bfrag[ni], (short)0, acc[mi][ni],
              false, false);
    }
  }

  // ---- epilogue: bias + leaky relu, store bf16
#pragma unroll
  for (int mi = 0; mi < 2; ++mi)
#pragma unroll
    for (int ni = 0; ni < 2; ++ni) {
#pragma unroll
      for (int r = 0; r < 8; ++r) {
        int co = coBlk + wm * 32 + mi * 16 + r + (laneHi ? 8 : 0);
        int t  = t0 + wn * 32 + ni * 16 + lane15;
        float v = acc[mi][ni][r] + bias[co];
        if (RELU) v = (v > 0.0f) ? v : 0.2f * v;
        out[((size_t)b * Cout + co) * (size_t)Tout + t] = (__bf16)v;
      }
    }
}

// ---------------------------------------------------------------------------
// conv3 (Cout=10, too small for matrix path) fused with mod_sigmoid(h-5).
//   h2 : [B][512][1024] bf16 -> amp : [B][512][2][5] float
// Block: 256 threads handle (b, 64 time outputs); 640 (co,t) pairs / block.
// ---------------------------------------------------------------------------
__global__ __launch_bounds__(256) void conv3_amp_kernel(
    const __bf16* __restrict__ h2, const float* __restrict__ w3,
    const float* __restrict__ b3, float* __restrict__ amp) {
  constexpr int CIN = 512, TIN = 1024;
  __shared__ __align__(16) __bf16 sX[132][32];
  __shared__ float sW[3][10][32];

  const int tid = threadIdx.x;
  const int t0  = blockIdx.x * 64;
  const int b   = blockIdx.y;

  float accv[3] = {0.f, 0.f, 0.f};

  for (int c0 = 0; c0 < CIN; c0 += 32) {
    __syncthreads();
    for (int idx = tid; idx < 10 * 32; idx += 256) {
      int co = idx >> 5, ci = idx & 31;
      const float* wp = w3 + ((size_t)co * CIN + (c0 + ci)) * 3;
      sW[0][co][ci] = wp[0];
      sW[1][co][ci] = wp[1];
      sW[2][co][ci] = wp[2];
    }
    for (int idx = tid; idx < 32 * 132; idx += 256) {
      int ci = idx / 132, d = idx % 132;
      int tg = 2 * t0 + d;
      float v = 0.0f;
      if (d < 130 && tg < TIN)
        v = (float)h2[((size_t)b * CIN + (c0 + ci)) * (size_t)TIN + tg];
      sX[d][ci] = (__bf16)v;
    }
    __syncthreads();
#pragma unroll
    for (int s = 0; s < 3; ++s) {
      int p = tid + s * 256;
      if (p < 640) {
        int co = p / 64, t = p % 64;
        float sum = accv[s];
#pragma unroll
        for (int tap = 0; tap < 3; ++tap)
          for (int ci = 0; ci < 32; ++ci)
            sum += (float)sX[2 * t + tap][ci] * sW[tap][co][ci];
        accv[s] = sum;
      }
    }
  }
#pragma unroll
  for (int s = 0; s < 3; ++s) {
    int p = tid + s * 256;
    if (p < 640) {
      int co = p / 64, t = p % 64;
      float h  = accv[s] + b3[co] - 5.0f;
      float sg = 1.0f / (1.0f + __expf(-h));
      float av = 2.0f * __powf(sg, 2.3f) + 1e-7f;
      int dd = co / 5, band = co % 5;
      amp[(((size_t)b * 512 + (t0 + t)) * 2 + dd) * 5 + band] = av;
    }
  }
}

// ---------------------------------------------------------------------------
// Tail: amp[5] -> ir (irfft+roll+hann+roll, closed form) -> causal conv with
// (2*noise-1). One thread per (b, t', d).  out : [B][2][T'*8] float
// ---------------------------------------------------------------------------
__global__ __launch_bounds__(256) void noise_tail_kernel(
    const float* __restrict__ amp, const float* __restrict__ noise,
    float* __restrict__ out) {
  int idx = blockIdx.x * 256 + threadIdx.x;
  if (idx >= 16 * 512 * 2) return;
  int d = idx & 1;
  int t = (idx >> 1) & 511;
  int b = idx >> 10;

  const float* ap = amp + (size_t)idx * 5;
  float a0 = ap[0], a1 = ap[1], a2 = ap[2], a3 = ap[3], a4 = ap[4];

  // C1[j] = cos(2*pi*j/8)
  const float C1[8] = {1.f, 0.70710678f, 0.f, -0.70710678f,
                       -1.f, -0.70710678f, 0.f, 0.70710678f};
  float ir[8];
#pragma unroll
  for (int n = 0; n < 8; ++n) {
    int m = (n + 4) & 7;   // roll(-4) of the windowed, roll(+4)'d irfft
    float c1 = C1[m];
    float c2 = C1[(2 * m) & 7];
    float c3 = C1[(3 * m) & 7];
    float c4 = (m & 1) ? -1.f : 1.f;
    float ir2 = 0.125f * (a0 + 2.f * (-a1 * c1 + a2 * c2 - a3 * c3) + a4 * c4);
    float wn  = 0.5f * (1.f - C1[m]);   // periodic Hann
    ir[n] = ir2 * wn;
  }

  const float* np = noise + (size_t)idx * 8;
  float nz[8];
#pragma unroll
  for (int j = 0; j < 8; ++j) nz[j] = 2.f * np[j] - 1.f;

  // fft_convolve(pad) + back half == causal truncated convolution
  float* op = out + ((size_t)b * 2 + d) * 4096 + (size_t)t * 8;
#pragma unroll
  for (int p = 0; p < 8; ++p) {
    float s = 0.f;
#pragma unroll
    for (int j = 0; j <= p; ++j) s += nz[j] * ir[p - j];
    op[p] = s;
  }
}

// ---------------------------------------------------------------------------
extern "C" void kernel_launch(void* const* d_in, const int* in_sizes, int n_in,
                              void* d_out, int out_size, void* d_ws,
                              size_t ws_size, hipStream_t stream) {
  const float* x     = (const float*)d_in[0];   // [16][512][4096]
  const float* noise = (const float*)d_in[1];   // [16][512][2][8]
  const float* w1    = (const float*)d_in[2];
  const float* b1    = (const float*)d_in[3];
  const float* w2    = (const float*)d_in[4];
  const float* b2    = (const float*)d_in[5];
  const float* w3    = (const float*)d_in[6];
  const float* b3    = (const float*)d_in[7];
  float* out = (float*)d_out;

  char* ws = (char*)d_ws;
  const size_t wb_bytes = (size_t)3 * 512 * 512 * sizeof(__bf16);    // 1.5 MB
  const size_t h1_bytes = (size_t)16 * 512 * 2048 * sizeof(__bf16);  // 32 MB
  const size_t h2_bytes = (size_t)16 * 512 * 1024 * sizeof(__bf16);  // 16 MB
  __bf16* wb1 = (__bf16*)ws;
  __bf16* wb2 = (__bf16*)(ws + wb_bytes);
  __bf16* h1  = (__bf16*)(ws + 2 * wb_bytes);
  __bf16* h2  = (__bf16*)(ws + 2 * wb_bytes + h1_bytes);
  float*  amp = (float*)(ws + 2 * wb_bytes + h1_bytes + h2_bytes);   // 320 KB

  // pre-transpose weights to bf16 [3][Cout][Cin]
  weight_prep_kernel<<<(512 * 512 + 255) / 256, 256, 0, stream>>>(w1, wb1, 512,
                                                                  512);
  weight_prep_kernel<<<(512 * 512 + 255) / 256, 256, 0, stream>>>(w2, wb2, 512,
                                                                  512);
  // conv1: 512->512, T 4096->2048, leaky relu
  conv_wmma_kernel<float, true><<<dim3(2048 / 64, 512 / 128, 16), 256, 0,
                                  stream>>>(x, wb1, b1, h1, 512, 512, 4096,
                                            2048);
  // conv2: 512->512, T 2048->1024, leaky relu
  conv_wmma_kernel<__bf16, true><<<dim3(1024 / 64, 512 / 128, 16), 256, 0,
                                   stream>>>(h1, wb2, b2, h2, 512, 512, 2048,
                                             1024);
  // conv3 + mod_sigmoid -> amp [B][512][2][5]
  conv3_amp_kernel<<<dim3(512 / 64, 16), 256, 0, stream>>>(h2, w3, b3, amp);
  // spectral-noise tail -> out [16][2][4096]
  noise_tail_kernel<<<(16 * 512 * 2 + 255) / 256, 256, 0, stream>>>(amp, noise,
                                                                    out);
}
